// BF_Model_82368882802800
// MI455X (gfx1250) — matrix operand order repaired and checked by reading, not compiled, not run on hardware
//
#include <hip/hip_runtime.h>
#include <stdint.h>

#define N_NODES 100000
#define N_EDGES 1600000
#define HID 64
#define N_LAYERS 2

typedef __attribute__((ext_vector_type(16))) _Float16 v16h;
typedef __attribute__((ext_vector_type(8)))  float    v8f;

// ---- monotone orderable encoding for fp32 min via u32 atomicMin -------------
__device__ __forceinline__ unsigned fenc(float f) {
  unsigned u = __float_as_uint(f);
  return (u & 0x80000000u) ? ~u : (u | 0x80000000u);
}
__device__ __forceinline__ float fdec(unsigned u) {
  unsigned v = (u & 0x80000000u) ? (u & 0x7FFFFFFFu) : ~u;
  return __uint_as_float(v);
}

// ---- fill agg with the "empty segment" sentinel (max encodable) -------------
__global__ void init_agg(unsigned* __restrict__ agg, int n) {
  int i = blockIdx.x * blockDim.x + threadIdx.x;
  int stride = gridDim.x * blockDim.x;
  for (; i < n; i += stride) agg[i] = 0xFFFFFFFFu;
}

// ---- edge kernel: per-edge MLP via WMMA + scatter-min atomics ---------------
// Each wave handles a tile of 16 edges. A = relu([x_src, ea] @ aW1 + ab1) is
// computed directly in the 16-bit WMMA A-matrix register layout with packed
// f16 FMA/MAX (v_pk_fma_f16 / v_pk_max_num_f16); B (aW2) is preloaded once per
// wave in the B layout; msg = A @ B (+ab2) via v_wmma_f32_16x16x32_f16, then
// encoded scatter-min to agg[dst].
__global__ void __launch_bounds__(256) edge_msg_min(
    const float* __restrict__ x,        // [N]
    const float* __restrict__ ea,       // [E]
    const long long* __restrict__ eidx, // [2,E]  (row0 = src, row1 = dst)
    const float* __restrict__ aW1,      // [2,64]
    const float* __restrict__ ab1,      // [64]
    const float* __restrict__ aW2,      // [64,64]
    const float* __restrict__ ab2,      // [64]
    unsigned* __restrict__ agg)         // [N,64] encoded
{
  const int lane = threadIdx.x & 31;
  const int m    = lane & 15;   // column / row-in-tile id
  const int hi   = lane >> 4;   // half-wave select

  // ---- preload B = aW2 in WMMA B layout: 4 N-tiles x 2 K-chunks ----
  // B operand (16-bit, 32x16), contiguous k order (same as A): lanes 0-15 hold K=0..15, lanes 16-31 hold
  // K=16..31; element i of v16h -> K = chunk*32 + hi*16 + i; column N = lane%16.
  v16h B[2][4];
#pragma unroll
  for (int c = 0; c < 2; ++c)
#pragma unroll
    for (int t = 0; t < 4; ++t) {
      const int n = t * 16 + m;
#pragma unroll
      for (int i = 0; i < 16; ++i) {
        const int k = c * 32 + hi * 16 + i;
        B[c][t][i] = (_Float16)aW2[k * HID + n];
      }
    }

  // ---- preload layer-1 weights in the A layout ----
  // A operand (16-bit, 16x32), contiguous k order (same as B): M = lane%16; element i -> K = chunk*32 + hi*16 +
  // i.
  v16h w0p[2], w1p[2], b1p[2];
#pragma unroll
  for (int c = 0; c < 2; ++c) {
#pragma unroll
    for (int i = 0; i < 16; ++i) {
      const int k = c * 32 + hi * 16 + i;
      w0p[c][i] = (_Float16)aW1[0 * HID + k];
      w1p[c][i] = (_Float16)aW1[1 * HID + k];
      b1p[c][i] = (_Float16)ab1[k];
    }
  }

  float bias2[4];
#pragma unroll
  for (int t = 0; t < 4; ++t) bias2[t] = ab2[t * 16 + m];

  const v16h zeroh = (v16h)(_Float16)0;

  const int numTiles = N_EDGES / 16;          // exact: 100000 tiles
  const int wave   = blockIdx.x * (blockDim.x >> 5) + (threadIdx.x >> 5);
  const int nWaves = gridDim.x * (blockDim.x >> 5);

  // wave-uniform loop -> EXEC stays all-ones around WMMA
  for (int tile = wave; tile < numTiles; tile += nWaves) {
    const int e = tile * 16 + m;
    const long long srcI = eidx[e];
    const int       dstI = (int)eidx[N_EDGES + e];
    const _Float16 xh = (_Float16)x[srcI];
    const _Float16 eh = (_Float16)ea[e];

    // h = relu(x*w0 + ea*w1 + b1), in the A register layout, as packed f16 ops
    v16h A[2];
#pragma unroll
    for (int c = 0; c < 2; ++c) {
      v16h h = w0p[c] * xh + w1p[c] * eh + b1p[c];  // v_pk_fma_f16
      A[c] = __builtin_elementwise_max(h, zeroh);    // v_pk_max_num_f16
    }

#pragma unroll
    for (int t = 0; t < 4; ++t) {
      v8f C = {};
      C = __builtin_amdgcn_wmma_f32_16x16x32_f16(false, A[0], false, B[0][t],
                                                 (short)0, C, false, false);
      C = __builtin_amdgcn_wmma_f32_16x16x32_f16(false, A[1], false, B[1][t],
                                                 (short)0, C, false, false);
      const int n = t * 16 + m;
#pragma unroll
      for (int r = 0; r < 8; ++r) {
        const int mrow = r + hi * 8;                  // edge row of C[r]
        const int d    = __shfl(dstI, mrow, 32);      // dst of that edge
        const float val = C[r] + bias2[t];
        atomicMin(&agg[(size_t)d * HID + n], fenc(val));
      }
    }
  }
}

// ---- node update MLP: x' = relu(agg @ uW1 + ub1) @ uW2 + ub2 ---------------
// Weights are wave-uniform (scalar loads / L2 broadcast); 0.8 GF/layer.
__global__ void __launch_bounds__(256) update_nodes(
    const unsigned* __restrict__ agg,  // [N,64] encoded
    const float* __restrict__ uW1,     // [64,64]
    const float* __restrict__ ub1,     // [64]
    const float* __restrict__ uW2,     // [64]
    const float* __restrict__ ub2,     // [1]
    float* __restrict__ xout, int n)
{
  const int node = blockIdx.x * blockDim.x + threadIdx.x;
  if (node >= n) return;

  float a[HID];
  const unsigned* row = agg + (size_t)node * HID;
#pragma unroll
  for (int k = 0; k < HID; ++k) {
    const unsigned u = row[k];
    float v = (u == 0xFFFFFFFFu) ? 0.0f : fdec(u);
    if (isinf(v)) v = 0.0f;          // reference: isinf(agg) -> 0
    a[k] = v;
  }

  float out = ub2[0];
  for (int j = 0; j < HID; ++j) {
    float s = ub1[j];
#pragma unroll
    for (int k = 0; k < HID; ++k) s = fmaf(a[k], uW1[k * HID + j], s);
    out = fmaf(fmaxf(s, 0.0f), uW2[j], out);
  }
  xout[node] = out;
}

extern "C" void kernel_launch(void* const* d_in, const int* in_sizes, int n_in,
                              void* d_out, int out_size, void* d_ws, size_t ws_size,
                              hipStream_t stream) {
  const float* x0  = (const float*)d_in[0];
  const float* ea  = (const float*)d_in[1];
  const float* aW1 = (const float*)d_in[2];   // [L,2,64]
  const float* ab1 = (const float*)d_in[3];   // [L,64]
  const float* aW2 = (const float*)d_in[4];   // [L,64,64]
  const float* ab2 = (const float*)d_in[5];   // [L,64]
  const float* uW1 = (const float*)d_in[6];   // [L,64,64]
  const float* ub1 = (const float*)d_in[7];   // [L,64]
  const float* uW2 = (const float*)d_in[8];   // [L,64,1]
  const float* ub2 = (const float*)d_in[9];   // [L,1]
  const long long* eidx = (const long long*)d_in[10]; // [2,E] int64

  // workspace: encoded agg buffer + intermediate node features
  unsigned* agg = (unsigned*)d_ws;
  float* x1 = (float*)((char*)d_ws + (size_t)N_NODES * HID * sizeof(unsigned));

  for (int l = 0; l < N_LAYERS; ++l) {
    const float* xin  = (l == 0) ? x0 : x1;
    float*       xout = (l == N_LAYERS - 1) ? (float*)d_out : x1;

    init_agg<<<1024, 256, 0, stream>>>(agg, N_NODES * HID);

    edge_msg_min<<<2048, 256, 0, stream>>>(
        xin, ea, eidx,
        aW1 + (size_t)l * 2 * HID, ab1 + (size_t)l * HID,
        aW2 + (size_t)l * HID * HID, ab2 + (size_t)l * HID, agg);

    update_nodes<<<(N_NODES + 255) / 256, 256, 0, stream>>>(
        agg,
        uW1 + (size_t)l * HID * HID, ub1 + (size_t)l * HID,
        uW2 + (size_t)l * HID, ub2 + l, xout, N_NODES);
  }
}
